// DownConvFace_12790412607766
// MI455X (gfx1250) — compile-verified, matrix operand order checked by reading, not run on hardware
//
#include <hip/hip_runtime.h>
#include <hip/hip_bf16.h>
#include <math.h>

// ---------------- types ----------------
typedef __attribute__((ext_vector_type(16))) __bf16 v16bf;
typedef __attribute__((ext_vector_type(8)))  float  v8f;
typedef __attribute__((ext_vector_type(8)))  unsigned short us8;

union Frag16 {           // 32 bytes: one WMMA bf16 A/B fragment per lane
    v16bf bf;
    us8   u[2];
};

static __device__ __forceinline__ unsigned short f2bf(float x) {
    // round-to-nearest-even float -> bf16
    unsigned int u = __float_as_uint(x);
    u += 0x7FFFu + ((u >> 16) & 1u);
    return (unsigned short)(u >> 16);
}

#define COUT 64
#define BATCH 4

// ---------------------------------------------------------------------------
// Mesh face conv as a WMMA GEMM:
//   D[64, B*F] = W_bf16[64, 4*CIN] x G_bf16[4*CIN, B*F] + bias
// G built on the fly from irregular neighbor gathers, staged in a
// double-buffered per-wave LDS tile. Block: 256 threads = 8 waves; each wave
// owns 16 faces x all 64 output channels (4 M-tiles of v8f accumulators).
// CIN is a template parameter so the K loop fully unrolls.
// ---------------------------------------------------------------------------
template <int CIN>
__global__ __launch_bounds__(256)
void mesh_conv_wmma_kernel(const float* __restrict__ X,      // [B, CIN, F]
                           const int*   __restrict__ gemm,   // [B, F, 3]
                           const float* __restrict__ W,      // [64, CIN, 4]
                           const float* __restrict__ bias,   // [64]
                           float*       __restrict__ Y,      // [B, 64, F]
                           int F)
{
    constexpr int K       = 4 * CIN;       // 128 or 256
    constexpr int NCHUNKS = K >> 5;        // 4 or 8

    // Weights staged once per block: [o][kk] bf16, row stride K halfs.
    __shared__ unsigned short wsh[COUT * K];
    // Per-wave double-buffered G tile: 2 x (16 faces x 40-half rows).
    __shared__ unsigned short gsh[8 * 2 * 16 * 40];
    __shared__ float bsh[COUT];

    const int tid  = threadIdx.x;
    const int wv   = tid >> 5;
    const int lane = tid & 31;
    const int b    = blockIdx.y;

    for (int i = tid; i < COUT * K; i += 256)
        wsh[i] = f2bf(W[i]);
    if (tid < COUT) bsh[tid] = bias[tid];
    __syncthreads();

    const int fbase = blockIdx.x * 128 + wv * 16;
    if (fbase >= F) return;                // wave-uniform exit (after barrier)

    const int face = lane & 15;
    const int hi16 = lane >> 4;            // 0 for lanes 0-15, 1 for 16-31

    int fg = fbase + face;
    if (fg >= F) fg = F - 1;               // clamp: keep EXEC all-1s for WMMA
    const int* gp = gemm + ((size_t)b * F + fg) * 3;
    int i1 = gp[0], i2 = gp[1], i3 = gp[2];
    i1 = min(max(i1, 0), F - 1);
    i2 = min(max(i2, 0), F - 1);
    i3 = min(max(i3, 0), F - 1);

    const float* Xb = X + (size_t)b * CIN * F;
    const int gwave = wv * (2 * 16 * 40);

    v8f acc[4];
    #pragma unroll
    for (int mt = 0; mt < 4; ++mt) acc[mt] = (v8f)0.0f;

    const us8* wrow = (const us8*)wsh;

    #pragma unroll
    for (int chunk = 0; chunk < NCHUNKS; ++chunk) {
        const int c0    = chunk * 8;
        const int gbase = gwave + (chunk & 1) * (16 * 40);
        const us8* grow = (const us8*)(gsh + gbase);

        // Prefetch next chunk's gather cachelines (same face indices, +8 channels).
        if (chunk + 1 < NCHUNKS) {
            const float* xn = Xb + (size_t)(c0 + 8 + hi16 * 4) * F;
            __builtin_prefetch(xn + fg, 0, 3);
            __builtin_prefetch(xn + i1, 0, 3);
            __builtin_prefetch(xn + i2, 0, 3);
            __builtin_prefetch(xn + i3, 0, 3);
        }

        // ---- build 16x32 bf16 G tile in LDS (this wave only, lockstep) ----
        #pragma unroll
        for (int it = 0; it < 4; ++it) {
            const int crel = it * 2 + hi16;          // 0..7 relative channel
            const float* xc = Xb + (size_t)(c0 + crel) * F;
            float ctr = xc[fg];
            float a1 = xc[i1], a2 = xc[i2], a3 = xc[i3];
            float fsum = a1 + a2 + a3;
            float fad  = fabsf(a1 - a2) + fabsf(a2 - a3) + fabsf(a3 - a1);
            float fmx  = fmaxf(a1, fmaxf(a2, a3));
            unsigned int lo = (unsigned int)f2bf(ctr) | ((unsigned int)f2bf(fsum) << 16);
            unsigned int hi = (unsigned int)f2bf(fad) | ((unsigned int)f2bf(fmx)  << 16);
            uint2* dst = (uint2*)&gsh[gbase + face * 40 + crel * 4];
            *dst = make_uint2(lo, hi);               // ds_store_b64
        }
        // ---- B fragment: lane n=face, K runs {0..7,16..23} / {8..15,24..31}
        Frag16 bf;
        {
            int rb = face * 5 + hi16;                // units of 8 halfs (16B)
            bf.u[0] = grow[rb];
            bf.u[1] = grow[rb + 2];
        }
        // ---- A fragments per M tile + WMMA accumulate ----
        #pragma unroll
        for (int mt = 0; mt < 4; ++mt) {
            Frag16 af;
            int m  = mt * 16 + face;
            int ka = chunk * 32 + (hi16 << 3);       // in halfs
            int ai = (m * K + ka) >> 3;              // units of 8 halfs
            af.u[0] = wrow[ai];
            af.u[1] = wrow[ai + 2];
            acc[mt] = __builtin_amdgcn_wmma_f32_16x16x32_bf16(
                false, af.bf, false, bf.bf, (short)0, acc[mt], false, false);
        }
    }

    // ---- epilogue: bias add + store (C/D layout: VGPR r -> M=r(+8), lane -> N)
    const int fg2 = fbase + face;
    float* yb = Y + (size_t)b * COUT * F;
    if (fg2 < F) {
        #pragma unroll
        for (int mt = 0; mt < 4; ++mt) {
            const int obase = mt * 16 + (hi16 << 3);
            #pragma unroll
            for (int r = 0; r < 8; ++r) {
                const int o = obase + r;
                yb[(size_t)o * F + fg2] = acc[mt][r] + bsh[o];
            }
        }
    }
}

// ---------------------------------------------------------------------------
// Per-channel sum / sum-of-squares over (B, F) for training-mode BatchNorm.
// One blockIdx.y per channel; inner loop over faces, batches unrolled.
// ---------------------------------------------------------------------------
__global__ __launch_bounds__(256)
void bn_stats_kernel(const float* __restrict__ Y, float* __restrict__ stats,
                     int F)
{
    const int o = blockIdx.y;
    const int tid = threadIdx.x;
    float s = 0.f, s2 = 0.f;
    for (int f = blockIdx.x * blockDim.x + tid; f < F;
         f += gridDim.x * blockDim.x) {
        #pragma unroll
        for (int bb = 0; bb < BATCH; ++bb) {
            float v = Y[((size_t)(bb * COUT + o)) * F + f];
            s += v; s2 += v * v;
        }
    }
    #pragma unroll
    for (int off = 16; off > 0; off >>= 1) {
        s  += __shfl_down(s,  off, 32);
        s2 += __shfl_down(s2, off, 32);
    }
    __shared__ float ps[8], ps2[8];
    const int lane = tid & 31, wv = tid >> 5;
    if (lane == 0) { ps[wv] = s; ps2[wv] = s2; }
    __syncthreads();
    if (tid < 32) {
        s  = (tid < 8) ? ps[tid]  : 0.f;
        s2 = (tid < 8) ? ps2[tid] : 0.f;
        #pragma unroll
        for (int off = 4; off > 0; off >>= 1) {
            s  += __shfl_down(s,  off, 32);
            s2 += __shfl_down(s2, off, 32);
        }
        if (tid == 0) {
            atomicAdd(&stats[o], s);
            atomicAdd(&stats[COUT + o], s2);
        }
    }
}

// ---------------------------------------------------------------------------
// Fused BN (training mode) + optional residual + LeakyReLU.
// blockIdx.y = b*64 + o  ->  per-channel params are block-uniform (s_loads).
// ---------------------------------------------------------------------------
__global__ __launch_bounds__(256)
void bn_act_kernel(const float* __restrict__ Y, const float* __restrict__ stats,
                   const float* __restrict__ gamma, const float* __restrict__ beta,
                   const float* __restrict__ resid, float* __restrict__ out,
                   int F, float invCount)
{
    const int bo = blockIdx.y;
    const int o  = bo & (COUT - 1);
    const float mean = stats[o] * invCount;
    const float var  = stats[COUT + o] * invCount - mean * mean;
    const float scl  = gamma[o] * rsqrtf(var + 1e-5f);
    const float sft  = beta[o] - mean * scl;

    const int f = blockIdx.x * blockDim.x + threadIdx.x;
    if (f >= F) return;
    const size_t i = (size_t)bo * F + f;
    float v = Y[i] * scl + sft;
    if (resid) v += resid[i];
    out[i] = (v >= 0.f) ? v : 0.2f * v;
}

__global__ void zero_stats_kernel(float* stats)
{
    if (threadIdx.x < 2 * COUT) stats[threadIdx.x] = 0.f;
}

// ---------------------------------------------------------------------------
extern "C" void kernel_launch(void* const* d_in, const int* in_sizes, int n_in,
                              void* d_out, int out_size, void* d_ws, size_t ws_size,
                              hipStream_t stream)
{
    const float* fe   = (const float*)d_in[0];
    const int*   gemm = (const int*)  d_in[1];
    const float* w1   = (const float*)d_in[2];
    const float* b1   = (const float*)d_in[3];
    const float* w2a  = (const float*)d_in[4];
    const float* b2a  = (const float*)d_in[5];
    const float* w2b  = (const float*)d_in[6];
    const float* b2b  = (const float*)d_in[7];
    const float* g0   = (const float*)d_in[8];
    const float* be0  = (const float*)d_in[9];
    const float* g1   = (const float*)d_in[10];
    const float* be1  = (const float*)d_in[11];
    const float* g2   = (const float*)d_in[12];
    const float* be2  = (const float*)d_in[13];

    const int B    = BATCH;
    const int Cin1 = in_sizes[2] / (COUT * 4);     // 32
    const int F    = in_sizes[0] / (B * Cin1);     // 50000
    const int n    = B * COUT * F;
    const float invCount = 1.0f / (float)(B * F);

    float* xbuf  = (float*)d_ws;                   // activations / residual
    float* ybuf  = xbuf + (size_t)n;               // raw conv output
    float* stats = ybuf + (size_t)n;               // [2*64] sums

    const dim3 cgrid((F + 127) / 128, B);
    const dim3 sgrid(64, COUT);
    const dim3 egrid((F + 255) / 256, B * COUT);

    // ---- stage 1: conv1 -> BN -> LeakyReLU --------------------------------
    if (Cin1 == 32)
        mesh_conv_wmma_kernel<32><<<cgrid, 256, 0, stream>>>(fe, gemm, w1, b1, ybuf, F);
    else
        mesh_conv_wmma_kernel<64><<<cgrid, 256, 0, stream>>>(fe, gemm, w1, b1, ybuf, F);
    zero_stats_kernel<<<1, 128, 0, stream>>>(stats);
    bn_stats_kernel<<<sgrid, 256, 0, stream>>>(ybuf, stats, F);
    bn_act_kernel<<<egrid, 256, 0, stream>>>(ybuf, stats, g0, be0, nullptr, xbuf,
                                             F, invCount);

    // ---- stage 2: residual block 1 ----------------------------------------
    mesh_conv_wmma_kernel<64><<<cgrid, 256, 0, stream>>>(xbuf, gemm, w2a, b2a, ybuf, F);
    zero_stats_kernel<<<1, 128, 0, stream>>>(stats);
    bn_stats_kernel<<<sgrid, 256, 0, stream>>>(ybuf, stats, F);
    bn_act_kernel<<<egrid, 256, 0, stream>>>(ybuf, stats, g1, be1, xbuf, xbuf,
                                             F, invCount);

    // ---- stage 3: residual block 2 -> d_out -------------------------------
    mesh_conv_wmma_kernel<64><<<cgrid, 256, 0, stream>>>(xbuf, gemm, w2b, b2b, ybuf, F);
    zero_stats_kernel<<<1, 128, 0, stream>>>(stats);
    bn_stats_kernel<<<sgrid, 256, 0, stream>>>(ybuf, stats, F);
    bn_act_kernel<<<egrid, 256, 0, stream>>>(ybuf, stats, g2, be2, xbuf,
                                             (float*)d_out, F, invCount);
}
